// RGCNNet_31086973288561
// MI455X (gfx1250) — matrix operand order repaired and checked by reading, not compiled, not run on hardware
//
#include <hip/hip_runtime.h>

// Problem constants (match reference)
#define Nn   40000
#define Ee   512000
#define Dd   128
#define Hh   128
#define Rr   8
#define Bb   8
#define Cc   3
#define KAGG 1024   // R*H  (relation-aggregated part of A)
#define KTOT 1152   // R*H + H (plus root/self part)

typedef __attribute__((ext_vector_type(16))) __bf16 v16bf;
typedef __attribute__((ext_vector_type(8)))  float  v8f;

union ABf16 {
    v16bf v;
    __bf16 e[16];
    uint4  q[2];
};

// ---------------------------------------------------------------- utilities
__global__ void zero4_kernel(float4* __restrict__ p, int n4) {
    int i = blockIdx.x * blockDim.x + threadIdx.x;
    if (i < n4) p[i] = make_float4(0.f, 0.f, 0.f, 0.f);
}

// per-(node,relation) in-degree (edge structure is layer-invariant)
__global__ void count_kernel(const int* __restrict__ dst, const int* __restrict__ et,
                             float* __restrict__ cnt) {
    int e = blockIdx.x * blockDim.x + threadIdx.x;
    if (e < Ee) atomicAdd(&cnt[dst[e] * Rr + et[e]], 1.0f);
}

// cnt -> 1/max(cnt,1)   (hoists the divide out of the GEMM inner loop)
__global__ void rcnt_kernel(float* __restrict__ cnt) {
    int i = blockIdx.x * blockDim.x + threadIdx.x;
    if (i < Nn * Rr) cnt[i] = 1.0f / fmaxf(cnt[i], 1.0f);
}

// h0 = prelu(num_x @ lin_w + lin_b, a0) + x
__global__ void init_h_kernel(const float* __restrict__ num_x, const float* __restrict__ x,
                              const float* __restrict__ lin_w, const float* __restrict__ lin_b,
                              const float* __restrict__ pa0, float* __restrict__ h) {
    int i = blockIdx.x * blockDim.x + threadIdx.x;   // N*D threads, exact
    int n = i >> 7, d = i & 127;
    float v = num_x[n] * lin_w[d] + lin_b[d];
    v = v >= 0.f ? v : pa0[d] * v;
    h[i] = v + x[i];
}

// one wave per edge: agg[(dst*R+etype), :] += h[src, :]
__global__ void scatter_kernel(const float* __restrict__ h, const int* __restrict__ src,
                               const int* __restrict__ dst, const int* __restrict__ et,
                               float* __restrict__ agg) {
    int e    = blockIdx.x * 8 + (threadIdx.x >> 5);
    int lane = threadIdx.x & 31;
    if (e >= Ee) return;
    int s = src[e], d = dst[e], t = et[e];
    const float* hs = h + (size_t)s * Hh;
    float* ab = agg + ((size_t)d * Rr + t) * (size_t)Hh;
#pragma unroll
    for (int j = 0; j < 4; ++j) {
        int c = lane + 32 * j;
        atomicAdd(&ab[c], hs[c]);
    }
}

// Build combined weight [KTOT, outCols] in bf16, stored COLUMN-major
// (W[col*KTOT + k]) so a B-fragment is two contiguous 16-byte loads per lane.
__global__ void build_w_kernel(const float* __restrict__ bases, const float* __restrict__ comp,
                               const float* __restrict__ root, unsigned short* __restrict__ W,
                               int outCols, int colsPad) {
    int i = blockIdx.x * blockDim.x + threadIdx.x;
    int total = colsPad * KTOT;
    if (i >= total) return;
    int row = i % KTOT;
    int col = i / KTOT;
    float s = 0.f;
    if (col < outCols) {
        if (row < KAGG) {
            int r = row >> 7, ii = row & 127;
#pragma unroll
            for (int b = 0; b < Bb; ++b)
                s += comp[r * Bb + b] * bases[((size_t)b * Hh + ii) * outCols + col];
        } else {
            s = root[(size_t)(row - KAGG) * outCols + col];
        }
    }
    __bf16 bf = (__bf16)s;
    W[(size_t)col * KTOT + row] = *(unsigned short*)&bf;
}

// ------------------------------------------------------------- WMMA GEMM
// A-fragment loader: lane holds row; j=0..7 -> K=base+hh*8+j, j=8..15 -> K=base+16+hh*8+j-8
__device__ __forceinline__ void loadA(ABf16& a, const float* __restrict__ p, int hh, float s) {
    const float4* p1 = (const float4*)(p + hh * 8);
    const float4* p2 = (const float4*)(p + 16 + hh * 8);
    float4 f0 = p1[0], f1 = p1[1], f2 = p2[0], f3 = p2[1];
    a.e[0]  = (__bf16)(f0.x * s); a.e[1]  = (__bf16)(f0.y * s);
    a.e[2]  = (__bf16)(f0.z * s); a.e[3]  = (__bf16)(f0.w * s);
    a.e[4]  = (__bf16)(f1.x * s); a.e[5]  = (__bf16)(f1.y * s);
    a.e[6]  = (__bf16)(f1.z * s); a.e[7]  = (__bf16)(f1.w * s);
    a.e[8]  = (__bf16)(f2.x * s); a.e[9]  = (__bf16)(f2.y * s);
    a.e[10] = (__bf16)(f2.z * s); a.e[11] = (__bf16)(f2.w * s);
    a.e[12] = (__bf16)(f3.x * s); a.e[13] = (__bf16)(f3.y * s);
    a.e[14] = (__bf16)(f3.z * s); a.e[15] = (__bf16)(f3.w * s);
}

#define WMMA_BF16(A, B, C) \
    __builtin_amdgcn_wmma_f32_16x16x32_bf16(false, (A), false, (B), (short)0, (C), false, false)

// out[m, col] = sum_k A[m,k] * W[k,col] + bias[col]  (optional PReLU)
// A[:,0:1024] = agg * rcnt (per-relation mean), A[:,1024:1152] = hcur.
// Each block: 32 output rows (two 16-row WMMA tiles sharing every B fragment).
// blockDim.x = 32 * nColTiles; wave w covers cols [w*16, w*16+16).
__global__ void rgcn_gemm_kernel(const float* __restrict__ agg, const float* __restrict__ rcnt,
                                 const float* __restrict__ hcur,
                                 const unsigned short* __restrict__ W,
                                 const float* __restrict__ bias, const float* __restrict__ pa,
                                 float* __restrict__ out, int outStride, int validCols) {
    int wave = threadIdx.x >> 5;
    int lane = threadIdx.x & 31;
    int m0   = blockIdx.x * 32;
    int o0   = wave * 16;
    int l15  = lane & 15;
    int hh   = lane >> 4;
    int ncol = o0 + l15;

    v8f c0 = {0.f, 0.f, 0.f, 0.f, 0.f, 0.f, 0.f, 0.f};
    v8f c1 = {0.f, 0.f, 0.f, 0.f, 0.f, 0.f, 0.f, 0.f};

    const float* arow0 = agg  + (size_t)(m0 + l15) * KAGG;
    const float* arow1 = arow0 + (size_t)16 * KAGG;
    const float* hrow0 = hcur + (size_t)(m0 + l15) * Hh;
    const float* hrow1 = hrow0 + (size_t)16 * Hh;
    const float* rrow0 = rcnt + (m0 + l15) * Rr;
    const float* rrow1 = rrow0 + 16 * Rr;
    const unsigned short* wcol = W + (size_t)ncol * KTOT;

    // relation-aggregated part: K = 0..1023, relation r owns K-chunk [r*128, r*128+128)
    for (int r = 0; r < Rr; ++r) {
        float rc0 = rrow0[r];
        float rc1 = rrow1[r];
        const float* a0p = arow0 + r * Hh;
        const float* a1p = arow1 + r * Hh;
        const unsigned short* wp = wcol + r * Hh;
        __builtin_prefetch(a0p + Hh, 0, 3);
        __builtin_prefetch(a1p + Hh, 0, 3);
#pragma unroll
        for (int kk = 0; kk < Hh; kk += 32) {
            ABf16 a0, a1, b;
            const uint4* bp = (const uint4*)(wp + kk + hh * 8);
            b.q[0] = bp[0];
            b.q[1] = bp[2];
            loadA(a0, a0p + kk, hh, rc0);
            loadA(a1, a1p + kk, hh, rc1);
            c0 = WMMA_BF16(a0.v, b.v, c0);
            c1 = WMMA_BF16(a1.v, b.v, c1);
        }
    }

    // root/self part: K = 1024..1151
#pragma unroll
    for (int kk = 0; kk < Hh; kk += 32) {
        ABf16 a0, a1, b;
        const uint4* bp = (const uint4*)(wcol + KAGG + kk + hh * 8);
        b.q[0] = bp[0];
        b.q[1] = bp[2];
        loadA(a0, hrow0 + kk, hh, 1.0f);
        loadA(a1, hrow1 + kk, hh, 1.0f);
        c0 = WMMA_BF16(a0.v, b.v, c0);
        c1 = WMMA_BF16(a1.v, b.v, c1);
    }

    // epilogue: C layout: lane (hh, n=l15), VGPR v -> row v + 8*hh (+16 for tile 1)
    if (ncol < validCols) {
        float bv = bias[ncol];
        float slope = (pa != nullptr) ? pa[ncol] : 1.0f;
#pragma unroll
        for (int v = 0; v < 8; ++v) {
            int m = m0 + v + hh * 8;
            float val0 = c0[v] + bv;
            float val1 = c1[v] + bv;
            if (pa != nullptr) {
                val0 = val0 >= 0.f ? val0 : slope * val0;
                val1 = val1 >= 0.f ? val1 : slope * val1;
            }
            out[(size_t)m * outStride + ncol]        = val0;
            out[(size_t)(m + 16) * outStride + ncol] = val1;
        }
    }
}

// log_softmax over 3 classes
__global__ void lsm_kernel(const float* __restrict__ logits, float* __restrict__ out) {
    int n = blockIdx.x * blockDim.x + threadIdx.x;
    if (n >= Nn) return;
    float a0 = logits[n * 16 + 0];
    float a1 = logits[n * 16 + 1];
    float a2 = logits[n * 16 + 2];
    float m = fmaxf(a0, fmaxf(a1, a2));
    float lse = m + logf(expf(a0 - m) + expf(a1 - m) + expf(a2 - m));
    out[n * 3 + 0] = a0 - lse;
    out[n * 3 + 1] = a1 - lse;
    out[n * 3 + 2] = a2 - lse;
}

// ---------------------------------------------------------------- launcher
extern "C" void kernel_launch(void* const* d_in, const int* in_sizes, int n_in,
                              void* d_out, int out_size, void* d_ws, size_t ws_size,
                              hipStream_t stream) {
    (void)in_sizes; (void)n_in; (void)out_size; (void)ws_size;

    const float* num_x  = (const float*)d_in[0];
    const float* x      = (const float*)d_in[1];
    const int*   eidx   = (const int*)d_in[2];
    const int*   etype  = (const int*)d_in[3];
    const float* lin_w  = (const float*)d_in[4];
    const float* lin_b  = (const float*)d_in[5];
    const float* prelu  = (const float*)d_in[6];   // [7, D]
    const float* bases  = (const float*)d_in[7];   // [6, B, H, H]
    const float* comp   = (const float*)d_in[8];   // [6, R, B]
    const float* root   = (const float*)d_in[9];   // [6, H, H]
    const float* bias   = (const float*)d_in[10];  // [6, H]
    const float* bases7 = (const float*)d_in[11];  // [B, H, C]
    const float* comp7  = (const float*)d_in[12];  // [R, B]
    const float* root7  = (const float*)d_in[13];  // [H, C]
    const float* bias7  = (const float*)d_in[14];  // [C]
    float* outp = (float*)d_out;

    const int* srcp = eidx;        // edge_index[0]
    const int* dstp = eidx + Ee;   // edge_index[1]

    // workspace carve-up
    char* ws = (char*)d_ws;
    size_t off = 0;
    float* h0  = (float*)(ws + off); off += (size_t)Nn * Hh * 4;
    float* h1  = (float*)(ws + off); off += (size_t)Nn * Hh * 4;
    float* agg = (float*)(ws + off); off += (size_t)Nn * Rr * Hh * 4;
    float* cnt = (float*)(ws + off); off += (size_t)Nn * Rr * 4;
    unsigned short* W = (unsigned short*)(ws + off); off += (size_t)128 * KTOT * 2;
    float* logits = (float*)(ws + off); off += (size_t)Nn * 16 * 4;

    // edge-count histogram -> reciprocal (layer invariant)
    zero4_kernel<<<(Nn * Rr / 4 + 255) / 256, 256, 0, stream>>>((float4*)cnt, Nn * Rr / 4);
    count_kernel<<<(Ee + 255) / 256, 256, 0, stream>>>(dstp, etype, cnt);
    rcnt_kernel<<<(Nn * Rr + 255) / 256, 256, 0, stream>>>(cnt);

    // input embedding
    init_h_kernel<<<Nn * Dd / 256, 256, 0, stream>>>(num_x, x, lin_w, lin_b, prelu, h0);

    float* hA = h0;
    float* hB = h1;
    for (int l = 0; l < 6; ++l) {
        build_w_kernel<<<(128 * KTOT + 255) / 256, 256, 0, stream>>>(
            bases + (size_t)l * Bb * Hh * Hh, comp + (size_t)l * Rr * Bb,
            root + (size_t)l * Hh * Hh, W, Hh, 128);
        zero4_kernel<<<(Nn * Rr * Hh / 4 + 255) / 256, 256, 0, stream>>>(
            (float4*)agg, Nn * Rr * Hh / 4);
        scatter_kernel<<<Ee / 8, 256, 0, stream>>>(hA, srcp, dstp, etype, agg);
        rgcn_gemm_kernel<<<Nn / 32, 256, 0, stream>>>(
            agg, cnt, hA, W, bias + (size_t)l * Hh, prelu + (size_t)(l + 1) * Dd,
            hB, Hh, Hh);
        float* t = hA; hA = hB; hB = t;
    }

    // conv7: 3 output classes, padded into one 16-col tile
    build_w_kernel<<<(16 * KTOT + 255) / 256, 256, 0, stream>>>(bases7, comp7, root7, W, Cc, 16);
    zero4_kernel<<<(Nn * Rr * Hh / 4 + 255) / 256, 256, 0, stream>>>(
        (float4*)agg, Nn * Rr * Hh / 4);
    scatter_kernel<<<Ee / 8, 256, 0, stream>>>(hA, srcp, dstp, etype, agg);
    rgcn_gemm_kernel<<<Nn / 32, 32, 0, stream>>>(agg, cnt, hA, W, bias7, nullptr,
                                                 logits, 16, Cc);

    lsm_kernel<<<(Nn + 255) / 256, 256, 0, stream>>>(logits, outp);
}
